// MambaActorNetwork_58059367907937
// MI455X (gfx1250) — matrix operand order repaired and checked by reading, not compiled, not run on hardware
//
#include <hip/hip_runtime.h>
#include <hip/hip_bf16.h>
#include <math.h>

#define B_  16
#define L_  2048
#define F_  64
#define H_  256
#define DI_ 512
#define M_  (B_ * L_)   // 32768 rows for all big GEMMs

typedef __attribute__((ext_vector_type(16))) _Float16     v16h;
typedef __attribute__((ext_vector_type(8)))  _Float16     v8h;
typedef __attribute__((ext_vector_type(4)))  _Float16     v4h;
typedef __attribute__((ext_vector_type(8)))  float        v8f;
typedef __attribute__((ext_vector_type(4)))  unsigned int v4u;
typedef __attribute__((ext_vector_type(8)))  int          v8i;
typedef __attribute__((ext_vector_type(4)))  int          v4i;

__device__ __forceinline__ float sigmoidf_(float x) { return 1.f / (1.f + __expf(-x)); }
__device__ __forceinline__ float siluf_(float x)    { return x * sigmoidf_(x); }

#if __has_builtin(__builtin_amdgcn_tensor_load_to_lds)
#define USE_TDM 1
#else
#define USE_TDM 0
#endif

// ---------------------------------------------------------------------------
// fp32 [Kd,N] -> f16 [N,Kd] transposed weight conversion (once per launch).
// Column-major weights make B-fragments K-contiguous in LDS.
// ---------------------------------------------------------------------------
__global__ __launch_bounds__(256)
void cvt_t_f16_kernel(const float* __restrict__ src, _Float16* __restrict__ dst,
                      int Kd, int N) {
    int i = blockIdx.x * 256 + threadIdx.x;     // i = n*Kd + k (coalesced write)
    if (i < Kd * N) {
        int n = i / Kd, k = i - n * Kd;
        dst[i] = (_Float16)src[(size_t)k * N + n];
    }
}

#if USE_TDM
// ---------------------------------------------------------------------------
// TDM: DMA a 128(n) x 32(k) f16 tile from WT[N][Kd] into LDS, with pad so
// each 64B row lands on an 80B LDS stride (conflict-free ds_load_b128 after).
// D# per CDNA5 ISA:
//   group0: count=1 | lds_addr | global_addr(57b) | type=2
//   group1: data_size=1(2B), pad_enable, pad_interval=3 (16 dwords),
//           pad_amount=3 (4 dwords), tensor_dim0=Kd, tensor_dim1=N,
//           tile_dim0=32, tile_dim1=128, tensor_dim0_stride=Kd
//   group2/3: zero (2D tile)
// ---------------------------------------------------------------------------
__device__ __forceinline__ void tdm_load_wt_tile(const _Float16* gsrc, unsigned lds_off, int Kd, int N)
{
    unsigned long long ga = (unsigned long long)(const void*)gsrc;
    v4u g0 = { 1u,                                   // count=1, user mode
               lds_off,                              // LDS byte address
               (unsigned)(ga & 0xFFFFFFFFu),         // global_addr[31:0]
               (unsigned)(((ga >> 32) & 0x01FFFFFFu) // global_addr[56:32]
                          | (2u << 30)) };           // type=2 ("image")
    v8i g1 = { (int)((1u << 16)                      // data_size=1 (2 bytes)
                     | (1u << 20)                    // pad_enable
                     | (3u << 22)                    // pad_interval: 16 dwords
                     | (3u << 25)),                  // pad_amount: 4 dwords
               (int)(((unsigned)Kd & 0xFFFFu) << 16),// tensor_dim0[15:0]=Kd
               (int)(((unsigned)N  & 0xFFFFu) << 16),// dim0 hi=0 | tensor_dim1[15:0]=N
               (int)(32u << 16),                     // dim1 hi=0 | tile_dim0=32
               128,                                  // tile_dim1=128 | tile_dim2=0
               Kd,                                   // tensor_dim0_stride[31:0]
               0, 0 };
    v4i z4 = { 0, 0, 0, 0 };
#if defined(__clang_major__) && (__clang_major__ >= 23)
    v8i z8 = { 0, 0, 0, 0, 0, 0, 0, 0 };
    __builtin_amdgcn_tensor_load_to_lds(g0, g1, z4, z4, z8, 0);
#else
    __builtin_amdgcn_tensor_load_to_lds(g0, g1, z4, z4, 0);
#endif
}
#endif

// ---------------------------------------------------------------------------
// WMMA GEMM: C[M,N] = epi(A[M,Kd] @ W[Kd,N] + bias[N]), W given transposed
// as WT[N,Kd] f16.  Block: 256 threads = 8 waves (2x4); wave tile 32x32 =
// 2x2 WMMA tiles; block tile 64x128; K stepped by 32.
// Requires: M % 64 == 0, N % 128 == 0, Kd % 32 == 0 (true for all uses).
// ---------------------------------------------------------------------------
enum { EPI_NONE = 0, EPI_SILU = 1, EPI_ALPHA = 2 };

template<int EPI>
__global__ __launch_bounds__(256)
void gemm_wmma_kernel(const float* __restrict__ A, const _Float16* __restrict__ WT,
                      const float* __restrict__ bias, float* __restrict__ C,
                      int Kd, int N)
{
    __shared__ _Float16 As[64 * 48];     // 64 rows x 32 k, stride 48 halves (96B)
    __shared__ _Float16 Wt[128 * 40];    // 128 cols x 32 k, stride 40 halves (80B)

    const int t     = threadIdx.x;
    const int lane  = t & 31;
    const int wave  = t >> 5;
    const int twRow = wave >> 2;          // 0..1 (32 rows each)
    const int twCol = wave & 3;           // 0..3 (32 cols each)
    const int half  = lane >> 4;
    const int ml    = lane & 15;
    const int gm0   = blockIdx.y * 64;
    const int gn0   = blockIdx.x * 128;

    const int aRow  = t >> 2;             // 0..63
    const int aCol8 = (t & 3) * 8;        // 0,8,16,24

    v8f acc[2][2] = {};
    const unsigned ldsW = (unsigned)(unsigned long long)(void*)&Wt[0];
    (void)ldsW;

    for (int k0 = 0; k0 < Kd; k0 += 32) {
        // ---- W tile: one wave kicks the Tensor Data Mover ----
#if USE_TDM
        if (wave == 0) tdm_load_wt_tile(WT + (size_t)gn0 * Kd + k0, ldsW, Kd, N);
#else
        {
            const int wr = t >> 1, wc16 = (t & 1) * 16;
            const _Float16* wp = WT + (size_t)(gn0 + wr) * Kd + k0 + wc16;
            *(v8h*)&Wt[wr * 40 + wc16]     = *(const v8h*)wp;
            *(v8h*)&Wt[wr * 40 + wc16 + 8] = *(const v8h*)(wp + 8);
        }
#endif
        // ---- A tile: fp32 -> f16 staging ----
        {
            const float* ap = A + (size_t)(gm0 + aRow) * Kd + k0 + aCol8;
            float4 a0 = *(const float4*)ap;
            float4 a1 = *(const float4*)(ap + 4);
            v4h h0 = { (_Float16)a0.x, (_Float16)a0.y, (_Float16)a0.z, (_Float16)a0.w };
            v4h h1 = { (_Float16)a1.x, (_Float16)a1.y, (_Float16)a1.z, (_Float16)a1.w };
            *(v4h*)&As[aRow * 48 + aCol8]     = h0;
            *(v4h*)&As[aRow * 48 + aCol8 + 4] = h1;
            if (k0 + 32 < Kd) __builtin_prefetch(ap + 32, 0, 1);  // global_prefetch_b8
        }
#if USE_TDM
        if (wave == 0) __builtin_amdgcn_s_wait_tensorcnt(0);
#endif
        __syncthreads();

        // ---- fragments (wave32 16x16x32 layout) ----
        v16h af[2], bf[2];
        #pragma unroll
        for (int s = 0; s < 2; ++s) {
            const int r = twRow * 32 + s * 16 + ml;
            v8h a0 = *(const v8h*)&As[r * 48 + 8 * half];       // K = 8h..8h+7
            v8h a1 = *(const v8h*)&As[r * 48 + 16 + 8 * half];  // K = 16+8h..+7
            #pragma unroll
            for (int j = 0; j < 8; ++j) { af[s][j] = a0[j]; af[s][8 + j] = a1[j]; }

            const int cn = twCol * 32 + s * 16 + ml;            // K-contiguous now
            v8h b0 = *(const v8h*)&Wt[cn * 40 + 8 * half];      // K = 8h..8h+7
            v8h b1 = *(const v8h*)&Wt[cn * 40 + 16 + 8 * half]; // K = 16+8h..+7
            #pragma unroll
            for (int j = 0; j < 8; ++j) { bf[s][j] = b0[j]; bf[s][8 + j] = b1[j]; }
        }
        #pragma unroll
        for (int sr = 0; sr < 2; ++sr)
            #pragma unroll
            for (int sc = 0; sc < 2; ++sc)
                acc[sr][sc] = __builtin_amdgcn_wmma_f32_16x16x32_f16(
                    false, af[sr], false, bf[sc], (short)0, acc[sr][sc], false, false);
        __syncthreads();
    }

    // ---- epilogue: D layout VGPR i -> row i + 8*half, col = ml ----
    #pragma unroll
    for (int sc = 0; sc < 2; ++sc) {
        const int col  = gn0 + twCol * 32 + sc * 16 + ml;
        const float bv = bias[col];
        #pragma unroll
        for (int sr = 0; sr < 2; ++sr) {
            #pragma unroll
            for (int i = 0; i < 8; ++i) {
                const int row = gm0 + twRow * 32 + sr * 16 + i + 8 * half;
                float z = acc[sr][sc][i] + bv;
                if (EPI == EPI_SILU) {
                    z = siluf_(z);
                } else if (EPI == EPI_ALPHA) {      // alpha = sigmoid(softplus(z))
                    float sp = (z > 20.f) ? z : log1pf(__expf(z));
                    z = sigmoidf_(sp);
                }
                C[(size_t)row * N + col] = z;
            }
        }
    }
}

// ---------------------------------------------------------------------------
// LayerNorm over rows of 256: out = [relu]( LN(x [+res]) * g + b )
// one wave (32 lanes) per row, 8 rows per 256-thread block
// ---------------------------------------------------------------------------
template<bool HAS_RES, bool RELU>
__global__ __launch_bounds__(256)
void ln256_kernel(const float* __restrict__ x, const float* __restrict__ res,
                  const float* __restrict__ g, const float* __restrict__ bb,
                  float* __restrict__ out)
{
    const int wave = threadIdx.x >> 5, lane = threadIdx.x & 31;
    const size_t row = (size_t)blockIdx.x * 8 + wave;
    const int cb = lane * 8;

    float v[8];
    {
        const float* xp = x + row * H_ + cb;
        float4 v0 = *(const float4*)xp;
        float4 v1 = *(const float4*)(xp + 4);
        v[0]=v0.x; v[1]=v0.y; v[2]=v0.z; v[3]=v0.w;
        v[4]=v1.x; v[5]=v1.y; v[6]=v1.z; v[7]=v1.w;
    }
    if (HAS_RES) {
        const float* rp = res + row * H_ + cb;
        float4 r0 = *(const float4*)rp;
        float4 r1 = *(const float4*)(rp + 4);
        v[0]+=r0.x; v[1]+=r0.y; v[2]+=r0.z; v[3]+=r0.w;
        v[4]+=r1.x; v[5]+=r1.y; v[6]+=r1.z; v[7]+=r1.w;
    }

    float s = 0.f;
    #pragma unroll
    for (int j = 0; j < 8; ++j) s += v[j];
    #pragma unroll
    for (int m = 16; m >= 1; m >>= 1) s += __shfl_xor(s, m, 32);
    const float mu = s * (1.f / H_);

    float q = 0.f;
    #pragma unroll
    for (int j = 0; j < 8; ++j) { float d = v[j] - mu; q += d * d; }
    #pragma unroll
    for (int m = 16; m >= 1; m >>= 1) q += __shfl_xor(q, m, 32);
    const float inv = rsqrtf(q * (1.f / H_) + 1e-5f);

    #pragma unroll
    for (int j = 0; j < 8; ++j) {
        float o = (v[j] - mu) * inv * g[cb + j] + bb[cb + j];
        if (RELU) o = fmaxf(o, 0.f);
        out[row * H_ + cb + j] = o;
    }
}

// ---------------------------------------------------------------------------
// depthwise causal conv K=4 + SiLU.  xr[M,1024] cols 0..511 already silu(xs).
// ---------------------------------------------------------------------------
__global__ __launch_bounds__(256)
void dwconv_kernel(const float* __restrict__ xr, const float* __restrict__ cw,
                   const float* __restrict__ cb, float* __restrict__ xs2)
{
    const size_t idx = (size_t)blockIdx.x * 256 + threadIdx.x;   // M_*DI_ threads
    const int    c   = (int)(idx % DI_);
    const size_t m   = idx / DI_;
    const int    l   = (int)(m % L_);

    float acc = cb[c];
    #pragma unroll
    for (int j = 0; j < 4; ++j) {
        const int ll = l - 3 + j;
        if (ll >= 0) acc += cw[c * 4 + j] * xr[(m - (size_t)(3 - j)) * 1024 + c];
    }
    xs2[idx] = siluf_(acc);
}

// ---------------------------------------------------------------------------
// selective scan along L, one lane per (batch, channel); in-place on xs2:
//   y_l = a_l*x_l + (1-a_l)*y_{l-1} (y_0 = x_0);  out = (y + x*D[c]) * silu(res)
// ---------------------------------------------------------------------------
__global__ __launch_bounds__(256)
void scan_kernel(float* __restrict__ xs2, const float* __restrict__ alpha,
                 const float* __restrict__ xr, const float* __restrict__ D)
{
    const int tid = blockIdx.x * 256 + threadIdx.x;   // 0..8191
    const int b = tid >> 9;
    const int c = tid & 511;
    const float d = D[c];

    float carry = 0.f;
    for (int l = 0; l < L_; ++l) {
        const size_t m   = (size_t)b * L_ + l;
        const size_t idx = m * DI_ + c;
        const float x = xs2[idx];
        const float a = alpha[idx];
        const float y = (l == 0) ? x : (a * x + (1.f - a) * carry);
        carry = y;
        const float r = xr[m * 1024 + 512 + c];       // silu(res) already applied
        xs2[idx] = (y + x * d) * r;
    }
}

// ---------------------------------------------------------------------------
// head: pf -> concat -> f1(LN,relu) -> f2(LN,relu) -> mean / clip(log_std)
// single block, 256 threads (M=16: too small for WMMA tiling to matter)
// ---------------------------------------------------------------------------
__global__ __launch_bounds__(256)
void head_kernel(const float* __restrict__ h,   const float* __restrict__ ps,
                 const float* __restrict__ pfW, const float* __restrict__ pfb,
                 const float* __restrict__ f1W, const float* __restrict__ f1b,
                 const float* __restrict__ f1g, const float* __restrict__ f1bn,
                 const float* __restrict__ f2W, const float* __restrict__ f2b,
                 const float* __restrict__ f2g, const float* __restrict__ f2bn,
                 const float* __restrict__ mW,  const float* __restrict__ mb,
                 const float* __restrict__ sW,  const float* __restrict__ sb,
                 float* __restrict__ outp)
{
    __shared__ float sc[16][320];
    __shared__ float sf[16][256];
    __shared__ float smu[16], sinv[16];
    const int t = threadIdx.x;

    // market features = h[:, L-1, :]
    for (int i = t; i < 16 * 256; i += 256) {
        const int b = i >> 8, j = i & 255;
        sc[b][j] = h[((size_t)b * L_ + (L_ - 1)) * H_ + j];
    }
    // pf = relu(ps @ pf_W + pf_b)
    for (int i = t; i < 16 * 64; i += 256) {
        const int b = i >> 6, j = i & 63;
        float v = ps[b * 2 + 0] * pfW[j] + ps[b * 2 + 1] * pfW[64 + j] + pfb[j];
        sc[b][256 + j] = fmaxf(v, 0.f);
    }
    __syncthreads();

    // f1: 320 -> 256
    for (int b = 0; b < 16; ++b) {
        float acc = f1b[t];
        for (int k = 0; k < 320; ++k) acc += sc[b][k] * f1W[k * 256 + t];
        sf[b][t] = acc;
    }
    __syncthreads();
    if (t < 16) {
        float s = 0.f;
        for (int k = 0; k < 256; ++k) s += sf[t][k];
        const float mu = s * (1.f / 256.f);
        float q = 0.f;
        for (int k = 0; k < 256; ++k) { float d = sf[t][k] - mu; q += d * d; }
        smu[t] = mu; sinv[t] = rsqrtf(q * (1.f / 256.f) + 1e-5f);
    }
    __syncthreads();
    for (int b = 0; b < 16; ++b)
        sf[b][t] = fmaxf((sf[b][t] - smu[b]) * sinv[b] * f1g[t] + f1bn[t], 0.f);
    __syncthreads();

    // f2: 256 -> 256 (output into sc[:, 0:256])
    for (int b = 0; b < 16; ++b) {
        float acc = f2b[t];
        for (int k = 0; k < 256; ++k) acc += sf[b][k] * f2W[k * 256 + t];
        sc[b][t] = acc;
    }
    __syncthreads();
    if (t < 16) {
        float s = 0.f;
        for (int k = 0; k < 256; ++k) s += sc[t][k];
        const float mu = s * (1.f / 256.f);
        float q = 0.f;
        for (int k = 0; k < 256; ++k) { float d = sc[t][k] - mu; q += d * d; }
        smu[t] = mu; sinv[t] = rsqrtf(q * (1.f / 256.f) + 1e-5f);
    }
    __syncthreads();
    for (int b = 0; b < 16; ++b)
        sc[b][t] = fmaxf((sc[b][t] - smu[b]) * sinv[b] * f2g[t] + f2bn[t], 0.f);
    __syncthreads();

    if (t < 16) {
        float m = mb[0], s = sb[0];
        for (int k = 0; k < 256; ++k) { m += sc[t][k] * mW[k]; s += sc[t][k] * sW[k]; }
        outp[t]      = m;
        outp[16 + t] = fminf(fmaxf(s, -2.f), 2.f);
    }
}

// ---------------------------------------------------------------------------
// launch
// ---------------------------------------------------------------------------
extern "C" void kernel_launch(void* const* d_in, const int* in_sizes, int n_in,
                              void* d_out, int out_size, void* d_ws, size_t ws_size,
                              hipStream_t stream)
{
    (void)in_sizes; (void)n_in; (void)out_size; (void)ws_size;

    // Input order: setup_inputs() insertion order, params flattened depth-first.
    // 0 market_data, 1 portfolio_state, then 4 blocks x
    //   {0 in_W,1 in_b,2 conv_w,3 conv_b,4 dt_W,5 dt_b,6 D,7 out_W,8 out_b,9 ln_g,10 ln_b},
    // 46 emb_W,47 emb_b,48 emb_g,49 emb_bn,50 pf_W,51 pf_b,52 f1_W,53 f1_b,
    // 54 f1_g,55 f1_bn,56 f2_W,57 f2_b,58 f2_g,59 f2_bn,60 m_W,61 m_b,62 s_W,63 s_b
    auto IN  = [&](int i) { return (const float*)d_in[i]; };
    auto BLK = [&](int blk, int j) { return (const float*)d_in[2 + 11 * blk + j]; };

    char*  ws  = (char*)d_ws;
    size_t off = 0;
    auto alloc = [&](size_t bytes) -> char* {
        char* p = ws + off;
        off += (bytes + 255) & ~((size_t)255);
        return p;
    };

    _Float16* embWT = (_Float16*)alloc((size_t)F_ * H_ * 2);
    _Float16 *inWT[4], *dtWT[4], *outWT[4];
    for (int b = 0; b < 4; ++b) {
        inWT[b]  = (_Float16*)alloc((size_t)H_  * (2 * DI_) * 2);
        dtWT[b]  = (_Float16*)alloc((size_t)DI_ * DI_ * 2);
        outWT[b] = (_Float16*)alloc((size_t)DI_ * H_  * 2);
    }
    float* bufH  = (float*)alloc((size_t)M_ * H_   * 4);
    float* bufXR = (float*)alloc((size_t)M_ * 1024 * 4);
    float* bufXS = (float*)alloc((size_t)M_ * DI_  * 4);
    float* bufAL = (float*)alloc((size_t)M_ * DI_  * 4);
    float* bufO  = (float*)alloc((size_t)M_ * H_   * 4);

    auto cvtT = [&](const float* s, _Float16* d, int Kd, int N) {
        int n = Kd * N;
        cvt_t_f16_kernel<<<(n + 255) / 256, 256, 0, stream>>>(s, d, Kd, N);
    };
    cvtT(IN(46), embWT, F_, H_);
    for (int b = 0; b < 4; ++b) {
        cvtT(BLK(b, 0), inWT[b],  H_,  2 * DI_);
        cvtT(BLK(b, 4), dtWT[b],  DI_, DI_);
        cvtT(BLK(b, 7), outWT[b], DI_, H_);
    }

    auto gemm = [&](int epi, const float* A, const _Float16* WT, const float* bias,
                    float* C, int Kd, int N) {
        dim3 grid(N / 128, M_ / 64);
        switch (epi) {
        case EPI_NONE:  gemm_wmma_kernel<EPI_NONE ><<<grid, 256, 0, stream>>>(A, WT, bias, C, Kd, N); break;
        case EPI_SILU:  gemm_wmma_kernel<EPI_SILU ><<<grid, 256, 0, stream>>>(A, WT, bias, C, Kd, N); break;
        default:        gemm_wmma_kernel<EPI_ALPHA><<<grid, 256, 0, stream>>>(A, WT, bias, C, Kd, N); break;
        }
    };

    // embedding + LN + relu
    gemm(EPI_NONE, IN(0), embWT, IN(47), bufO, F_, H_);
    ln256_kernel<false, true><<<M_ / 8, 256, 0, stream>>>(bufO, nullptr, IN(48), IN(49), bufH);

    // mamba blocks
    for (int b = 0; b < 4; ++b) {
        gemm(EPI_SILU, bufH, inWT[b], BLK(b, 1), bufXR, H_, 2 * DI_);
        dwconv_kernel<<<(size_t)M_ * DI_ / 256, 256, 0, stream>>>(bufXR, BLK(b, 2), BLK(b, 3), bufXS);
        gemm(EPI_ALPHA, bufXS, dtWT[b], BLK(b, 5), bufAL, DI_, DI_);
        scan_kernel<<<(B_ * DI_) / 256, 256, 0, stream>>>(bufXS, bufAL, bufXR, BLK(b, 6));
        gemm(EPI_NONE, bufXS, outWT[b], BLK(b, 8), bufO, DI_, H_);
        ln256_kernel<true, false><<<M_ / 8, 256, 0, stream>>>(bufO, bufH, BLK(b, 9), BLK(b, 10), bufH);
    }

    // head
    head_kernel<<<1, 256, 0, stream>>>(bufH, IN(1),
                                       IN(50), IN(51), IN(52), IN(53), IN(54), IN(55),
                                       IN(56), IN(57), IN(58), IN(59),
                                       IN(60), IN(61), IN(62), IN(63),
                                       (float*)d_out);
}
